// RelativeGlobalAttention_9225589752324
// MI455X (gfx1250) — matrix operand order repaired and checked by reading, not compiled, not run on hardware
//
#include <hip/hip_runtime.h>

typedef _Float16 v16h __attribute__((ext_vector_type(16)));
typedef _Float16 v8h  __attribute__((ext_vector_type(8)));
typedef _Float16 v4h  __attribute__((ext_vector_type(4)));
typedef float    v8f  __attribute__((ext_vector_type(8)));
typedef float    v4f  __attribute__((ext_vector_type(4)));

#define RB  4
#define RL  2048
#define RD  512
#define RH  8
#define RDH 64
#define RLT 2064   // padded column count for transposed V ([B,H,64,RLT])

// ---------------------------------------------------------------------------
// f32 -> f16 elementwise convert, 4 elements/thread (b128 in, b64 out)
// ---------------------------------------------------------------------------
__global__ __launch_bounds__(256) void f32_to_f16_kernel(const float* __restrict__ in,
                                                         _Float16* __restrict__ out, int n4) {
  for (int i = blockIdx.x * blockDim.x + threadIdx.x; i < n4; i += gridDim.x * blockDim.x) {
    v4f x = ((const v4f*)in)[i];
    v4h y;
#pragma unroll
    for (int j = 0; j < 4; ++j) y[j] = (_Float16)x[j];
    ((v4h*)out)[i] = y;
  }
}

// ---------------------------------------------------------------------------
// Fragment loaders (wave32, 16x16x32 f16 WMMA layouts)
// A (16x32, MxK), row-major source with K contiguous:
//   lane: row M = lane&15; halves 0..15 = K in [kbase+16*hf, +16) contiguous.
// B (32x16, KxN) from "Bt" = N-major storage (row n = B column n):
//   lane: col N = lane&15; halves 0..15 = K in [kbase+16*hf, +16) contiguous.
// ---------------------------------------------------------------------------
__device__ inline v16h frag_a_f16(const _Float16* __restrict__ base, int lda,
                                  int m0, int kbase, int lane) {
  int r = lane & 15, hf = lane >> 4;
  const _Float16* p = base + (size_t)(m0 + r) * lda + kbase + 16 * hf;
  v8h x = *(const v8h*)p;
  v8h y = *(const v8h*)(p + 8);
  v16h a;
#pragma unroll
  for (int i = 0; i < 8; ++i) { a[i] = x[i]; a[8 + i] = y[i]; }
  return a;
}

__device__ inline v16h frag_b_f16(const _Float16* __restrict__ p) {
  v8h x = *(const v8h*)p;
  v8h y = *(const v8h*)(p + 8);
  v16h b;
#pragma unroll
  for (int i = 0; i < 8; ++i) { b[i] = x[i]; b[8 + i] = y[i]; }
  return b;
}

// ---------------------------------------------------------------------------
// GEMM: out = X(8192x512,f16) @ W(512x512,f16)^T + bias(f32)
// One 16x64 tile per wave (A fragment reused across 4 N-subtiles).
// mode 0: f16 out, [B,H,L,DH] layout     (q, k projections)
// mode 1: f16 out, [B,H,DH,RLT] layout   (v projection, transposed, padded)
// mode 2: f32 out, [M,N] row-major       (final fc -> d_out)
// ---------------------------------------------------------------------------
__global__ __launch_bounds__(256) void gemm16_kernel(const _Float16* __restrict__ X,
                                                     const _Float16* __restrict__ W,
                                                     const float* __restrict__ bias,
                                                     void* __restrict__ out, int mode) {
  int lane = threadIdx.x & 31;
  int wid  = blockIdx.x * 8 + (threadIdx.x >> 5);
  int m0 = (wid >> 3) << 4;        // 512 M-tiles
  int ng = (wid & 7) << 6;         // N-group base: 0..448 step 64
  int ln = lane & 15, hf = lane >> 4;

  v8f c0 = {}, c1 = {}, c2 = {}, c3 = {};
#pragma unroll 2
  for (int kk = 0; kk < 16; ++kk) {
    int kbase = kk * 32;
    v16h a = frag_a_f16(X, RD, m0, kbase, lane);
    const _Float16* wp = W + (size_t)(ng + ln) * RD + kbase + 16 * hf;
    v16h b0 = frag_b_f16(wp);
    c0 = __builtin_amdgcn_wmma_f32_16x16x32_f16(false, a, false, b0, (short)0, c0, false, false);
    v16h b1 = frag_b_f16(wp + (size_t)16 * RD);
    c1 = __builtin_amdgcn_wmma_f32_16x16x32_f16(false, a, false, b1, (short)0, c1, false, false);
    v16h b2 = frag_b_f16(wp + (size_t)32 * RD);
    c2 = __builtin_amdgcn_wmma_f32_16x16x32_f16(false, a, false, b2, (short)0, c2, false, false);
    v16h b3 = frag_b_f16(wp + (size_t)48 * RD);
    c3 = __builtin_amdgcn_wmma_f32_16x16x32_f16(false, a, false, b3, (short)0, c3, false, false);
  }
#pragma unroll
  for (int j = 0; j < 4; ++j) {
    v8f c = (j == 0) ? c0 : (j == 1) ? c1 : (j == 2) ? c2 : c3;
    int n = ng + 16 * j + ln;            // global col = h*RDH + dh
    float bv = bias[n];
#pragma unroll
    for (int t = 0; t < 8; ++t) {
      int m = m0 + t + 8 * hf;           // global row = b*RL + l
      float val = c[t] + bv;
      if (mode == 0) {
        ((_Float16*)out)[(((size_t)(m >> 11) * RH + (n >> 6)) * RL + (m & (RL - 1))) * RDH + (n & 63)] =
            (_Float16)val;
      } else if (mode == 1) {
        ((_Float16*)out)[(((size_t)(m >> 11) * RH + (n >> 6)) * RDH + (n & 63)) * RLT + (m & (RL - 1))] =
            (_Float16)val;
      } else {
        ((float*)out)[(size_t)m * RD + n] = val;
      }
    }
  }
}

// ---------------------------------------------------------------------------
// Fused causal relative attention (flash style), one wave per (b,h, 16 rows).
// logits = (q.k^T + q.E[L-1-(l-m)]) / 8, causal mask -1e9, softmax, @ v.
// The 16x32 QE band at key-tile m0 overlaps the band at m0+16 by 16 columns,
// so the upper half-tile is carried in registers between iterations.
// ---------------------------------------------------------------------------
__global__ __launch_bounds__(256) void relattn_kernel(const _Float16* __restrict__ qp,
                                                      const _Float16* __restrict__ kp,
                                                      const _Float16* __restrict__ vt,
                                                      const _Float16* __restrict__ eh,
                                                      _Float16* __restrict__ aout) {
  __shared__ float lds_qe[8][16 * 32];
  __shared__ float lds_p[8][16 * 16];
  int lane  = threadIdx.x & 31;
  int wslot = threadIdx.x >> 5;
  int wid   = blockIdx.x * 8 + wslot;
  int bh = wid >> 7;                 // 0..31
  int l0 = (wid & 127) << 4;         // query tile start
  int b = bh >> 3, h = bh & 7;
  int ln = lane & 15, hf = lane >> 4;

  const _Float16* qb = qp + (size_t)bh * RL * RDH;
  const _Float16* kb = kp + (size_t)bh * RL * RDH;
  const _Float16* vb = vt + (size_t)bh * RDH * RLT;

  v16h aq0 = frag_a_f16(qb, RDH, l0, 0, lane);
  v16h aq1 = frag_a_f16(qb, RDH, l0, 32, lane);

  float mi[8], li[8];
  v8f acc0 = {}, acc1 = {}, acc2 = {}, acc3 = {};
#pragma unroll
  for (int t = 0; t < 8; ++t) { mi[t] = -3.0e38f; li[t] = 0.0f; }

  float* qe = lds_qe[wslot];
  float* pp = lds_p[wslot];

  // Prologue: QE carry tile for rows RL-16-l0+ln (always in-bounds), and zero
  // qe columns 16..31 (never rewritten; read as zero wherever N > M, on every key tile).
  v8f carry = {};
  {
    int row = RL - 16 - l0 + ln;
    v16h e0 = frag_b_f16(eh + (size_t)row * RDH + 16 * hf);
    v16h e1 = frag_b_f16(eh + (size_t)row * RDH + 32 + 16 * hf);
    carry = __builtin_amdgcn_wmma_f32_16x16x32_f16(false, aq0, false, e0, (short)0, carry, false, false);
    carry = __builtin_amdgcn_wmma_f32_16x16x32_f16(false, aq1, false, e1, (short)0, carry, false, false);
  }
#pragma unroll
  for (int t = 0; t < 8; ++t) qe[(t + 8 * hf) * 32 + 16 + ln] = 0.0f;

  for (int m0 = 0; m0 <= l0; m0 += 16) {
    // ---- S = q @ k^T  (K = 64) ----
    v8f S = {};
    {
      v16h b0 = frag_b_f16(kb + (size_t)(m0 + ln) * RDH + 16 * hf);
      S = __builtin_amdgcn_wmma_f32_16x16x32_f16(false, aq0, false, b0, (short)0, S, false, false);
      v16h b1 = frag_b_f16(kb + (size_t)(m0 + ln) * RDH + 32 + 16 * hf);
      S = __builtin_amdgcn_wmma_f32_16x16x32_f16(false, aq1, false, b1, (short)0, S, false, false);
    }
    // ---- QE band: columns 0..15 come from the carried tile ----
#pragma unroll
    for (int t = 0; t < 8; ++t) qe[(t + 8 * hf) * 32 + ln] = carry[t];
    if (m0 < l0) {
      // next carry: rows RL-(l0-m0)+ln, in-bounds since l0-m0 >= 16
      int row = RL - (l0 - m0) + ln;
      v16h e0 = frag_b_f16(eh + (size_t)row * RDH + 16 * hf);
      v16h e1 = frag_b_f16(eh + (size_t)row * RDH + 32 + 16 * hf);
      v8f cn = {};
      cn = __builtin_amdgcn_wmma_f32_16x16x32_f16(false, aq0, false, e0, (short)0, cn, false, false);
      cn = __builtin_amdgcn_wmma_f32_16x16x32_f16(false, aq1, false, e1, (short)0, cn, false, false);
      carry = cn;
    }
    // ---- add skewed rel, scale, mask ----
#pragma unroll
    for (int t = 0; t < 8; ++t) {
      int M = t + 8 * hf;
      float s = S[t] + qe[M * 32 + 15 + ln - M];   // t_idx = 15 + N - M in [0,30]
      s *= 0.125f;                                  // 1/sqrt(64)
      if (m0 + ln > l0 + M) s -= 1.0e9f;            // causal mask
      S[t] = s;
    }
    // ---- online softmax update ----
    float sc[8];
#pragma unroll
    for (int t = 0; t < 8; ++t) {
      float x = S[t];
      x = fmaxf(x, __shfl_xor(x, 1));
      x = fmaxf(x, __shfl_xor(x, 2));
      x = fmaxf(x, __shfl_xor(x, 4));
      x = fmaxf(x, __shfl_xor(x, 8));
      float mnew = fmaxf(mi[t], x);
      sc[t] = __expf(mi[t] - mnew);
      mi[t] = mnew;
      float p = __expf(S[t] - mnew);
      S[t] = p;
      float rs = p;
      rs += __shfl_xor(rs, 1);
      rs += __shfl_xor(rs, 2);
      rs += __shfl_xor(rs, 4);
      rs += __shfl_xor(rs, 8);
      li[t] = li[t] * sc[t] + rs;
    }
#pragma unroll
    for (int t = 0; t < 8; ++t) {
      acc0[t] *= sc[t]; acc1[t] *= sc[t]; acc2[t] *= sc[t]; acc3[t] *= sc[t];
    }
    // ---- P (C layout) -> LDS -> A fragment (K padded 16..31 with zeros) ----
#pragma unroll
    for (int t = 0; t < 8; ++t) pp[(t + 8 * hf) * 16 + ln] = S[t];
    v16h ap;
#pragma unroll
    for (int i = 0; i < 8; ++i) {
      ap[i]     = hf ? (_Float16)0.0f : (_Float16)pp[ln * 16 + i];
      ap[8 + i] = hf ? (_Float16)0.0f : (_Float16)pp[ln * 16 + 8 + i];
    }
    // ---- O += P @ V  (vT rows = output dims; K = key index, padded buf) ----
    {
      v16h bv0 = frag_b_f16(vb + (size_t)(0  + ln) * RLT + m0 + 16 * hf);
      acc0 = __builtin_amdgcn_wmma_f32_16x16x32_f16(false, ap, false, bv0, (short)0, acc0, false, false);
      v16h bv1 = frag_b_f16(vb + (size_t)(16 + ln) * RLT + m0 + 16 * hf);
      acc1 = __builtin_amdgcn_wmma_f32_16x16x32_f16(false, ap, false, bv1, (short)0, acc1, false, false);
      v16h bv2 = frag_b_f16(vb + (size_t)(32 + ln) * RLT + m0 + 16 * hf);
      acc2 = __builtin_amdgcn_wmma_f32_16x16x32_f16(false, ap, false, bv2, (short)0, acc2, false, false);
      v16h bv3 = frag_b_f16(vb + (size_t)(48 + ln) * RLT + m0 + 16 * hf);
      acc3 = __builtin_amdgcn_wmma_f32_16x16x32_f16(false, ap, false, bv3, (short)0, acc3, false, false);
    }
  }
  // ---- normalize + store [B,L,D] f16 ----
#pragma unroll
  for (int t = 0; t < 8; ++t) {
    int l = l0 + t + 8 * hf;
    float inv = 1.0f / li[t];
    size_t base = ((size_t)b * RL + l) * RD + (size_t)h * RDH;
    aout[base + 0  + ln] = (_Float16)(acc0[t] * inv);
    aout[base + 16 + ln] = (_Float16)(acc1[t] * inv);
    aout[base + 32 + ln] = (_Float16)(acc2[t] * inv);
    aout[base + 48 + ln] = (_Float16)(acc3[t] * inv);
  }
}

// ---------------------------------------------------------------------------
extern "C" void kernel_launch(void* const* d_in, const int* in_sizes, int n_in,
                              void* d_out, int out_size, void* d_ws, size_t ws_size,
                              hipStream_t stream) {
  (void)in_sizes; (void)n_in; (void)out_size; (void)ws_size;
  const float* Q    = (const float*)d_in[0];
  const float* K    = (const float*)d_in[1];
  const float* V    = (const float*)d_in[2];
  // d_in[3] = mask (implicit causal, unused)
  const float* Wq_w = (const float*)d_in[4];
  const float* Wq_b = (const float*)d_in[5];
  const float* Wk_w = (const float*)d_in[6];
  const float* Wk_b = (const float*)d_in[7];
  const float* Wv_w = (const float*)d_in[8];
  const float* Wv_b = (const float*)d_in[9];
  const float* fc_w = (const float*)d_in[10];
  const float* fc_b = (const float*)d_in[11];
  const float* E    = (const float*)d_in[12];

  char* ws = (char*)d_ws;
  size_t off = 0;
  auto alloc = [&](size_t bytes) {
    char* p = ws + off;
    off += (bytes + 255) & ~(size_t)255;
    return p;
  };
  const size_t NX = (size_t)RB * RL * RD;        // 4194304
  const size_t NW = (size_t)RD * RD;             // 262144
  const size_t NE = (size_t)RL * RDH;            // 131072

  _Float16* Qh  = (_Float16*)alloc(NX * 2);      // later reused as attn output
  _Float16* Kh  = (_Float16*)alloc(NX * 2);
  _Float16* Vh  = (_Float16*)alloc(NX * 2);
  _Float16* Wqh = (_Float16*)alloc(NW * 2);
  _Float16* Wkh = (_Float16*)alloc(NW * 2);
  _Float16* Wvh = (_Float16*)alloc(NW * 2);
  _Float16* Wfh = (_Float16*)alloc(NW * 2);
  _Float16* Eh  = (_Float16*)alloc(NE * 2);
  _Float16* qpj = (_Float16*)alloc(NX * 2);
  _Float16* kpj = (_Float16*)alloc(NX * 2);
  _Float16* vtb = (_Float16*)alloc((size_t)RB * RH * RDH * RLT * 2);

  f32_to_f16_kernel<<<1024, 256, 0, stream>>>(Q, Qh, (int)(NX / 4));
  f32_to_f16_kernel<<<1024, 256, 0, stream>>>(K, Kh, (int)(NX / 4));
  f32_to_f16_kernel<<<1024, 256, 0, stream>>>(V, Vh, (int)(NX / 4));
  f32_to_f16_kernel<<<128, 256, 0, stream>>>(Wq_w, Wqh, (int)(NW / 4));
  f32_to_f16_kernel<<<128, 256, 0, stream>>>(Wk_w, Wkh, (int)(NW / 4));
  f32_to_f16_kernel<<<128, 256, 0, stream>>>(Wv_w, Wvh, (int)(NW / 4));
  f32_to_f16_kernel<<<128, 256, 0, stream>>>(fc_w, Wfh, (int)(NW / 4));
  f32_to_f16_kernel<<<64, 256, 0, stream>>>(E, Eh, (int)(NE / 4));

  gemm16_kernel<<<512, 256, 0, stream>>>(Qh, Wqh, Wq_b, qpj, 0);
  gemm16_kernel<<<512, 256, 0, stream>>>(Kh, Wkh, Wk_b, kpj, 0);
  gemm16_kernel<<<512, 256, 0, stream>>>(Vh, Wvh, Wv_b, vtb, 1);

  relattn_kernel<<<512, 256, 0, stream>>>(qpj, kpj, vtb, Eh, Qh);

  gemm16_kernel<<<512, 256, 0, stream>>>(Qh, Wfh, fc_b, d_out, 2);
}
